// ExtractModel_11209864642693
// MI455X (gfx1250) — matrix operand order repaired and checked by reading, not compiled, not run on hardware
//
#include <hip/hip_runtime.h>
#include <math.h>

typedef __attribute__((ext_vector_type(16))) _Float16 v16h;
typedef __attribute__((ext_vector_type(8)))  _Float16 v8h;
typedef __attribute__((ext_vector_type(8)))  float    v8f;

#define B_    16
#define L_    64
#define D_    256
#define U_    500
#define UP_   512          // U padded to multiple of 16 for WMMA tiles
#define V_    10000
#define MINWL 4
#define MAXWL 10
#define WL_   7
#define M_    (B_ * L_)    // 1024 text positions
#define INV_T 10.0f        // 1 / TEMP, TEMP = 0.1

// ---------------------------------------------------------------------------
// Kernel 1: row-normalize to f16.  One block per row, 256 threads (D_ == 256).
// Rows >= valid_rows (the U->UP pad) are written as zeros.
// ---------------------------------------------------------------------------
__global__ void k_normalize(const float* __restrict__ in,
                            _Float16* __restrict__ out, int valid_rows) {
    __shared__ float red[256];
    const int row = blockIdx.x, t = threadIdx.x;
    float x = (row < valid_rows) ? in[row * D_ + t] : 0.0f;
    red[t] = x * x;
    __syncthreads();
    for (int s = 128; s > 0; s >>= 1) {
        if (t < s) red[t] += red[t + s];
        __syncthreads();
    }
    const float inv = (row < valid_rows) ? rsqrtf(red[0]) : 0.0f;
    out[row * D_ + t] = (_Float16)(x * inv);
}

// ---------------------------------------------------------------------------
// Kernel 2: sim[m,n] = sum_k A[m,k] * Bm[n,k]  via v_wmma_f32_16x16x32_f16.
// One wave (32 threads) per 16x16 tile; K=256 -> 8 chained WMMAs.
// A/B fragments use contiguous k = 16*half + i (same for both); f32 C/D per CDNA5 ISA 7.12.2.
// ---------------------------------------------------------------------------
__global__ void k_sim_wmma(const _Float16* __restrict__ A,
                           const _Float16* __restrict__ Bm,
                           float* __restrict__ C) {
    const int m0   = blockIdx.x * 16;
    const int n0   = blockIdx.y * 16;
    const int lane = threadIdx.x & 31;
    const int half = lane >> 4;    // K-half select
    const int l15  = lane & 15;    // M (for A) / N (for B/C)

    const _Float16* arow = A  + (size_t)(m0 + l15) * D_;
    const _Float16* brow = Bm + (size_t)(n0 + l15) * D_;

    v8f acc = {};
#pragma unroll
    for (int k0 = 0; k0 < D_; k0 += 32) {
        // A fragment: element i -> K = k0 + half*16 + i, M = l15
        v16h a = *(const v16h*)(arow + k0 + half * 16);
        // B fragment: element i -> K = k0 + half*16 + i, N = l15
        v16h b = *(const v16h*)(brow + k0 + half * 16);
        acc = __builtin_amdgcn_wmma_f32_16x16x32_f16(
            /*neg_a=*/false, a, /*neg_b=*/false, b,
            /*c_mod=*/(short)0, acc, /*reuse_a=*/false, /*reuse_b=*/false);
    }
    // C/D layout: VGPR r -> M = m0 + half*8 + r, N = n0 + l15
    const int mbase = m0 + half * 8;
#pragma unroll
    for (int r = 0; r < 8; ++r)
        C[(size_t)(mbase + r) * UP_ + n0 + l15] = acc[r];
}

// ---------------------------------------------------------------------------
// Kernel 3: per (b,s): async-stage 10 sim rows into LDS (ASYNCcnt path),
// stream V vocab entries with online soft-min (+argmin), then emit
// matched_ed / matched_vocab and the dense score row.
// 1024 blocks x 256 threads (8 waves).
// ---------------------------------------------------------------------------
__global__ void k_match(const float* __restrict__ sim,
                        const int* __restrict__ seg,
                        const int* __restrict__ vlen,
                        float* __restrict__ med,
                        int* __restrict__ mvoc,
                        float* __restrict__ dense) {
    __shared__ float lsim[MAXWL * UP_];                    // 20 KB
    __shared__ float rm[8][WL_], rz[8][WL_], rs[8][WL_], rv[8][WL_];
    __shared__ int   ri[8][WL_];

    const int bs = blockIdx.x;
    const int b  = bs >> 6;
    const int s  = bs & (L_ - 1);
    const int t  = threadIdx.x;

    // ---- Stage sim rows s..s+9 with GLOBAL_LOAD_ASYNC_TO_LDS_B128 ----
    // Wave-relative LDS byte offset == low 32 bits of the generic pointer.
    const unsigned lds_base = (unsigned)(size_t)lsim;
    int issued_async = 0;
    for (int q = t; q < MAXWL * (UP_ / 4); q += 256) {     // 1280 x 16B chunks
        const int j  = q >> 7;                             // row (128 vec4/row)
        const int l  = s + j;
        const unsigned lds_addr = lds_base + (unsigned)q * 16u;
        if (l < L_) {
            const unsigned long long gaddr =
                (unsigned long long)(size_t)&sim[(size_t)(b * L_ + l) * UP_ + (q & 127) * 4];
            asm volatile("global_load_async_to_lds_b128 %0, %1, off"
                         :: "v"(lds_addr), "v"(gaddr) : "memory");
            issued_async = 1;
        } else {
            // reference zero-padding of sim past L
            float4* p = (float4*)&lsim[q * 4];
            *p = make_float4(0.f, 0.f, 0.f, 0.f);
        }
    }
    (void)issued_async;
    asm volatile("s_wait_asynccnt 0" ::: "memory");
    __syncthreads();

    float m[WL_], Z[WL_], S[WL_], mv[WL_];
    int   mi[WL_];
#pragma unroll
    for (int k = 0; k < WL_; ++k) {
        m[k] = 3.4e38f; Z[k] = 0.f; S[k] = 0.f; mv[k] = 3.4e38f; mi[k] = 0x7fffffff;
    }

    for (int v = t; v < V_; v += 256) {
        if (v + 256 < V_)                                  // hide L2 latency of
            __builtin_prefetch(&seg[(v + 256) * MAXWL], 0, 3);  // next seg row
        const int len = vlen[v];
        float pf[MAXWL + 1];
        pf[0] = 0.0f;
#pragma unroll
        for (int j = 0; j < MAXWL; ++j) {
            const int u = seg[v * MAXWL + j];
            const float d = (s + j < L_) ? (1.0f - lsim[(j << 9) + u]) : 1.0f;
            pf[j + 1] = pf[j] + d;
        }
#pragma unroll
        for (int k = 0; k < WL_; ++k) {
            const int wl = MINWL + k;
            const int mm = wl < len ? wl : len;
            const float e = pf[mm] + fabsf((float)(wl - len));
            if (e < m[k]) {                    // new running min: rescale
                const float c = __expf((e - m[k]) * INV_T);
                Z[k] = Z[k] * c + 1.0f;
                S[k] = S[k] * c + e;
                m[k] = e;
            } else {
                const float w = __expf((m[k] - e) * INV_T);
                Z[k] += w;
                S[k] += e * w;
            }
            if (e < mv[k]) { mv[k] = e; mi[k] = v; }   // first-min (stride order)
        }
    }

    // wave32 butterfly reduction of the soft-min stats
#pragma unroll
    for (int k = 0; k < WL_; ++k) {
        for (int off = 16; off > 0; off >>= 1) {
            const float m2 = __shfl_xor(m[k],  off, 32);
            const float Z2 = __shfl_xor(Z[k],  off, 32);
            const float S2 = __shfl_xor(S[k],  off, 32);
            const float v2 = __shfl_xor(mv[k], off, 32);
            const int   i2 = __shfl_xor(mi[k], off, 32);
            const float mn = fminf(m[k], m2);
            const float c1 = __expf((mn - m[k]) * INV_T);
            const float c2 = __expf((mn - m2)  * INV_T);
            Z[k] = Z[k] * c1 + Z2 * c2;
            S[k] = S[k] * c1 + S2 * c2;
            m[k] = mn;
            if (v2 < mv[k] || (v2 == mv[k] && i2 < mi[k])) { mv[k] = v2; mi[k] = i2; }
        }
    }
    const int wave = t >> 5;
    if ((t & 31) == 0) {
#pragma unroll
        for (int k = 0; k < WL_; ++k) {
            rm[wave][k] = m[k]; rz[wave][k] = Z[k]; rs[wave][k] = S[k];
            rv[wave][k] = mv[k]; ri[wave][k] = mi[k];
        }
    }
    __syncthreads();

    // zero the dense row for this (b,s); scores overwrite below after a barrier
    if (t < L_) dense[(size_t)(b * L_ + s) * L_ + t] = 0.0f;

    float score_k = 0.0f;
    int   viable_k = 0;
    if (t < WL_) {
        const int k = t;
        float Mk = rm[0][k], Zk = rz[0][k], Sk = rs[0][k], Vk = rv[0][k];
        int   Ik = ri[0][k];
        for (int w = 1; w < 8; ++w) {
            const float mn = fminf(Mk, rm[w][k]);
            const float c1 = __expf((mn - Mk) * INV_T);
            const float c2 = __expf((mn - rm[w][k]) * INV_T);
            Zk = Zk * c1 + rz[w][k] * c2;
            Sk = Sk * c1 + rs[w][k] * c2;
            Mk = mn;
            if (rv[w][k] < Vk || (rv[w][k] == Vk && ri[w][k] < Ik)) {
                Vk = rv[w][k]; Ik = ri[w][k];
            }
        }
        const float matched = Sk / Zk;
        med [bs * WL_ + k] = matched;
        mvoc[bs * WL_ + k] = Ik;

        const int   wl = MINWL + k;
        const float y  = 1.0f - 2.0f * (matched / (float)wl);
        const float celu = (y > 0.0f) ? y : (__expf(y) - 1.0f);
        score_k  = 0.5f * (celu + 1.0f) * (float)wl;
        viable_k = (s + wl <= L_);
    }
    __syncthreads();
    if (t < WL_ && viable_k)
        dense[(size_t)(b * L_ + s) * L_ + (s + MINWL + t - 1)] = score_k;
}

// ---------------------------------------------------------------------------
// Kernel 4: per batch: online soft-max + argmax over 4096 dense spans,
// then decode (start, end, best_score, best_vocab).  16 blocks x 256.
// ---------------------------------------------------------------------------
__global__ void k_best(const float* __restrict__ dense,
                       const int* __restrict__ mvoc,
                       float* __restrict__ out) {
    __shared__ float rM[8], rZ[8], rS[8], rV[8];
    __shared__ int   rI[8];
    const int b = blockIdx.x, t = threadIdx.x;
    const float* f = dense + (size_t)b * L_ * L_;

    float M = -3.4e38f, Z = 0.f, S = 0.f, bv = -3.4e38f;
    int   bi = 0x7fffffff;
    for (int i = t; i < L_ * L_; i += 256) {
        const float x = f[i];
        if (x > M) {
            const float c = __expf((M - x) * INV_T);
            Z = Z * c + 1.0f; S = S * c + x; M = x;
        } else {
            const float w = __expf((x - M) * INV_T);
            Z += w; S += x * w;
        }
        if (x > bv) { bv = x; bi = i; }
    }
    for (int off = 16; off > 0; off >>= 1) {
        const float M2 = __shfl_xor(M, off, 32);
        const float Z2 = __shfl_xor(Z, off, 32);
        const float S2 = __shfl_xor(S, off, 32);
        const float v2 = __shfl_xor(bv, off, 32);
        const int   i2 = __shfl_xor(bi, off, 32);
        const float Mx = fmaxf(M, M2);
        const float c1 = __expf((M - Mx) * INV_T);
        const float c2 = __expf((M2 - Mx) * INV_T);
        Z = Z * c1 + Z2 * c2;
        S = S * c1 + S2 * c2;
        M = Mx;
        if (v2 > bv || (v2 == bv && i2 < bi)) { bv = v2; bi = i2; }
    }
    if ((t & 31) == 0) { const int w = t >> 5; rM[w] = M; rZ[w] = Z; rS[w] = S; rV[w] = bv; rI[w] = bi; }
    __syncthreads();
    if (t == 0) {
        M = rM[0]; Z = rZ[0]; S = rS[0]; bv = rV[0]; bi = rI[0];
        for (int w = 1; w < 8; ++w) {
            const float Mx = fmaxf(M, rM[w]);
            const float c1 = __expf((M - Mx) * INV_T);
            const float c2 = __expf((rM[w] - Mx) * INV_T);
            Z = Z * c1 + rZ[w] * c2;
            S = S * c1 + rS[w] * c2;
            M = Mx;
            if (rV[w] > bv || (rV[w] == bv && rI[w] < bi)) { bv = rV[w]; bi = rI[w]; }
        }
        const float best = S / Z;
        const int start = bi / L_;
        const int end   = bi % L_;
        int wsel = end - start + 1 - MINWL;
        wsel = wsel < 0 ? 0 : (wsel > WL_ - 1 ? WL_ - 1 : wsel);
        const int bvoc = mvoc[(b * L_ + start) * WL_ + wsel];
        out[b]          = (float)start;
        out[B_ + b]     = (float)end;
        out[2 * B_ + b] = best;
        out[3 * B_ + b] = (float)bvoc;
    }
}

// ---------------------------------------------------------------------------
extern "C" void kernel_launch(void* const* d_in, const int* in_sizes, int n_in,
                              void* d_out, int out_size, void* d_ws, size_t ws_size,
                              hipStream_t stream) {
    const float* word = (const float*)d_in[0];   // (B,L,D) f32
    const float* unit = (const float*)d_in[1];   // (U,D)   f32
    const int*   seg  = (const int*)d_in[2];     // (V,MAXWL) i32
    const int*   vlen = (const int*)d_in[3];     // (V,)      i32
    float*       out  = (float*)d_out;           // start,end,score,vocab,dense

    char* ws = (char*)d_ws;
    _Float16* wn16 = (_Float16*)ws;                                  // 1024*256 f16
    _Float16* un16 = (_Float16*)(ws + (size_t)M_ * D_ * 2);          // 512*256  f16
    float*    sim  = (float*)(ws + (size_t)M_ * D_ * 2 + (size_t)UP_ * D_ * 2);
    float*    med  = (float*)((char*)sim + (size_t)M_ * UP_ * 4);    // 1024*7 f32
    int*      mvoc = (int*)((char*)med + (size_t)M_ * WL_ * 4);      // 1024*7 i32

    float* dense = out + 4 * B_;                 // (B,L,L) region of d_out

    k_normalize<<<M_,  256, 0, stream>>>(word, wn16, M_);
    k_normalize<<<UP_, 256, 0, stream>>>(unit, un16, U_);
    k_sim_wmma<<<dim3(M_ / 16, UP_ / 16), 32, 0, stream>>>(wn16, un16, sim);
    k_match<<<M_, 256, 0, stream>>>(sim, seg, vlen, med, mvoc, dense);
    k_best<<<B_, 256, 0, stream>>>(dense, mvoc, out);
}
